// SparseMoELayer_57440892617409
// MI455X (gfx1250) — matrix operand order repaired and checked by reading, not compiled, not run on hardware
//
#include <hip/hip_runtime.h>

// Problem constants (match reference: B=8, N=8192, D=512, E=8, H=1024)
#define TOK  65536
#define DDIM 512
#define NEXP 8
#define HDIM 1024

#define MT   64      // token tile (M)
#define AS_S 520     // As LDS row stride (bf16 elems): 1040B -> 4-bank rotation
#define WS_S 40      // Ws LDS row stride: 80B -> conflict-free 16-lane access
#define HS_S 72      // Hs LDS row stride: 144B

typedef __bf16 bf16;
typedef bf16  v8bf  __attribute__((ext_vector_type(8)));
typedef bf16  v16bf __attribute__((ext_vector_type(16)));
typedef float v8f   __attribute__((ext_vector_type(8)));

__device__ __forceinline__ bf16 f2bf(float f) {
  unsigned u = __float_as_uint(f);
  u += 0x7FFFu + ((u >> 16) & 1u);            // round-to-nearest-even
  unsigned short s = (unsigned short)(u >> 16);
  return __builtin_bit_cast(bf16, s);
}

__device__ __forceinline__ v16bf cat8(v8bf lo, v8bf hi) {
  return __builtin_shufflevector(lo, hi, 0, 1, 2, 3, 4, 5, 6, 7,
                                         8, 9, 10, 11, 12, 13, 14, 15);
}

// A fragment (16x32 bf16, interleaved K): lane = (r, hi); elems 0..7 -> K=k0+8*hi+0..7,
// elems 8..15 -> K=k0+8*hi+16..23. Row pointer must be the lane's M row.
__device__ __forceinline__ v16bf load_a_frag(const bf16* row, int k0, int hi) {
  const bf16* p = row + k0 + 8 * hi;
  return cat8(*(const v8bf*)p, *(const v8bf*)(p + 16));
}

// B fragment (32x16 bf16, interleaved K): lane = (col r, hi);
// elems 0..7 -> K = 8*hi + 0..7, elems 8..15 -> K = 8*hi + 16..23. Row pointer is the lane's N row in the
// transposed LDS staging buffer.
__device__ __forceinline__ v16bf load_b_frag(const bf16* row, int hi) {
  const bf16* p = row + 8 * hi;
  return cat8(*(const v8bf*)p, *(const v8bf*)(p + 16));
}

__global__ void moe_init(int* __restrict__ counts) {
  if (threadIdx.x < NEXP) counts[threadIdx.x] = 0;
}

// One wave32 per token: logits = x_t . Wg + bg; top-1 argmax (first-max on ties,
// matching jnp.argmax); append token to its expert's list.
__global__ __launch_bounds__(256)
void moe_router(const float* __restrict__ x, const float* __restrict__ Wg,
                const float* __restrict__ bg, int* __restrict__ counts,
                int* __restrict__ lists) {
  const int wv   = threadIdx.x >> 5;
  const int lane = threadIdx.x & 31;
  const int t    = blockIdx.x * 8 + wv;              // grid.x = TOK/8 exactly

  const float* xr = x + (size_t)t * DDIM;
  float acc[NEXP];
#pragma unroll
  for (int e = 0; e < NEXP; ++e) acc[e] = 0.0f;

#pragma unroll 4
  for (int j = 0; j < DDIM / 32; ++j) {
    const int d = j * 32 + lane;
    const float xv = xr[d];                          // coalesced 128B per wave
    const float* wr = Wg + (size_t)d * NEXP;
#pragma unroll
    for (int e = 0; e < NEXP; ++e) acc[e] += xv * wr[e];
  }
#pragma unroll
  for (int e = 0; e < NEXP; ++e)
    for (int off = 16; off > 0; off >>= 1)
      acc[e] += __shfl_xor(acc[e], off, 32);

  if (lane == 0) {
    int bi = 0;
    float bv = acc[0] + bg[0];
#pragma unroll
    for (int e = 1; e < NEXP; ++e) {
      float v = acc[e] + bg[e];
      if (v > bv) { bv = v; bi = e; }                // strict > keeps first max
    }
    int pos = atomicAdd(&counts[bi], 1);
    lists[bi * TOK + pos] = t;
  }
}

// Fused expert MLP: y = relu(x W1[e] + b1[e]) W2[e] + b2[e] for this expert's
// token tile. 8 waves, 64-token tile, bf16 WMMA with f32 accumulation.
__global__ __launch_bounds__(256)
void moe_ffn(const float* __restrict__ x,
             const float* __restrict__ W1, const float* __restrict__ b1,
             const float* __restrict__ W2, const float* __restrict__ b2,
             const int* __restrict__ counts, const int* __restrict__ lists,
             float* __restrict__ out) {
  const int e    = blockIdx.y;
  const int cnt  = counts[e];
  const int base = blockIdx.x * MT;
  if (base >= cnt) return;                            // uniform early exit

  extern __shared__ char smem[];
  bf16* As = (bf16*)smem;                                             // 64 x 520
  bf16* Ws = (bf16*)(smem + (size_t)MT * AS_S * 2);                   // 512 x 40
  bf16* Hs = (bf16*)(smem + ((size_t)MT * AS_S + 512 * WS_S) * 2);    // 64 x 72
  int*  ti = (int*)(smem + ((size_t)MT * AS_S + 512 * WS_S + MT * HS_S) * 2);

  const int tid  = threadIdx.x;
  const int wv   = tid >> 5;
  const int lane = tid & 31;
  const int r    = lane & 15;
  const int hi   = lane >> 4;

  if (tid < MT) {
    const int g = base + tid;
    ti[tid] = (g < cnt) ? lists[e * TOK + g] : -1;
  }
  __syncthreads();

  // Gather token rows -> bf16 A tile (invalid rows zero-filled)
  for (int idx = tid; idx < MT * DDIM; idx += 256) {
    const int rr = idx >> 9;
    const int d  = idx & (DDIM - 1);
    const int t  = ti[rr];
    const float v = (t >= 0) ? x[(size_t)t * DDIM + d] : 0.0f;
    As[rr * AS_S + d] = f2bf(v);
  }

  const float* W1e = W1 + (size_t)e * DDIM * HDIM;
  const float* W2e = W2 + (size_t)e * HDIM * DDIM;
  const float* b1e = b1 + e * HDIM;
  const float* b2e = b2 + e * DDIM;

  const v8f vzero = {0.f, 0.f, 0.f, 0.f, 0.f, 0.f, 0.f, 0.f};
  v8f yacc[16];                                       // 64x64 y slice per wave
#pragma unroll
  for (int i = 0; i < 16; ++i) yacc[i] = vzero;

  for (int c = 0; c < HDIM / 64; ++c) {
    const int h0 = c * 64;
    v8f hacc0 = vzero, hacc1 = vzero;                 // 2 h-tiles per wave

    // ---- GEMM1: h_chunk(64x64) = A(64x512) * W1[:, h0:h0+64] ----
    for (int k0 = 0; k0 < DDIM; k0 += 32) {
      __syncthreads();                                // Ws free / A tile ready
      for (int idx = tid; idx < 32 * 64; idx += 256) {
        const int k = idx >> 6;
        const int n = idx & 63;
        Ws[n * WS_S + k] = f2bf(W1e[(size_t)(k0 + k) * HDIM + h0 + n]);
      }
      if (k0 + 32 < DDIM)
        __builtin_prefetch(&W1e[(size_t)(k0 + 32) * HDIM + h0], 0, 1);
      __syncthreads();

      const int t0  = wv * 2;
      const int mi0 = t0 >> 2, ni0 = t0 & 3;
      const int mi1 = (t0 + 1) >> 2, ni1 = (t0 + 1) & 3;
      v16bf a0 = load_a_frag(As + (mi0 * 16 + r) * AS_S, k0, hi);
      v16bf b0 = load_b_frag(Ws + (ni0 * 16 + r) * WS_S, hi);
      hacc0 = __builtin_amdgcn_wmma_f32_16x16x32_bf16(
          false, a0, false, b0, (short)0, hacc0, false, false);
      v16bf a1 = load_a_frag(As + (mi1 * 16 + r) * AS_S, k0, hi);
      v16bf b1f = load_b_frag(Ws + (ni1 * 16 + r) * WS_S, hi);
      hacc1 = __builtin_amdgcn_wmma_f32_16x16x32_bf16(
          false, a1, false, b1f, (short)0, hacc1, false, false);
    }

    // bias + ReLU -> Hs (bf16), C layout: m = i + 8*hi, n = r
#pragma unroll
    for (int tt = 0; tt < 2; ++tt) {
      const int tile = wv * 2 + tt;
      const int mi = tile >> 2, ni = tile & 3;
      const v8f hv = tt ? hacc1 : hacc0;
      const float bias = b1e[h0 + ni * 16 + r];
#pragma unroll
      for (int i = 0; i < 8; ++i) {
        float v = hv[i] + bias;
        v = v > 0.0f ? v : 0.0f;
        Hs[(mi * 16 + i + 8 * hi) * HS_S + ni * 16 + r] = f2bf(v);
      }
    }
    __syncthreads();                                  // Hs ready, GEMM1 reads done

    // ---- GEMM2: y(64x512) += h_chunk(64x64) * W2[h0:h0+64, :] ----
    for (int kk = 0; kk < 64; kk += 32) {
      for (int idx = tid; idx < 32 * 512; idx += 256) {
        const int k = idx >> 9;
        const int n = idx & 511;
        Ws[n * WS_S + k] = f2bf(W2e[(size_t)(h0 + kk + k) * DDIM + n]);
      }
      __syncthreads();
#pragma unroll
      for (int mi = 0; mi < 4; ++mi) {
        v16bf a = load_a_frag(Hs + (mi * 16 + r) * HS_S, kk, hi);
#pragma unroll
        for (int ni = 0; ni < 4; ++ni) {
          v16bf b = load_b_frag(Ws + (wv * 64 + ni * 16 + r) * WS_S, hi);
          yacc[mi * 4 + ni] = __builtin_amdgcn_wmma_f32_16x16x32_bf16(
              false, a, false, b, (short)0, yacc[mi * 4 + ni], false, false);
        }
      }
      __syncthreads();                                // before Ws reuse
    }
  }

  // Store y + b2 to routed token rows (f32, 16-lane coalesced segments)
#pragma unroll
  for (int mi = 0; mi < 4; ++mi) {
#pragma unroll
    for (int i = 0; i < 8; ++i) {
      const int m = mi * 16 + i + 8 * hi;
      if (base + m < cnt) {
        const int t = ti[m];
#pragma unroll
        for (int ni = 0; ni < 4; ++ni) {
          const int n = wv * 64 + ni * 16 + r;
          out[(size_t)t * DDIM + n] = yacc[mi * 4 + ni][i] + b2e[n];
        }
      }
    }
  }
}

extern "C" void kernel_launch(void* const* d_in, const int* in_sizes, int n_in,
                              void* d_out, int out_size, void* d_ws, size_t ws_size,
                              hipStream_t stream) {
  (void)in_sizes; (void)n_in; (void)out_size; (void)ws_size;
  const float* x  = (const float*)d_in[0];
  const float* W1 = (const float*)d_in[1];
  const float* b1 = (const float*)d_in[2];
  const float* W2 = (const float*)d_in[3];
  const float* b2 = (const float*)d_in[4];
  const float* Wg = (const float*)d_in[5];
  const float* bg = (const float*)d_in[6];
  float* out = (float*)d_out;

  int* counts = (int*)d_ws;           // 8 ints
  int* lists  = counts + 16;          // 8 * TOK ints (~2 MB)

  moe_init<<<1, 32, 0, stream>>>(counts);
  moe_router<<<TOK / 8, 256, 0, stream>>>(x, Wg, bg, counts, lists);

  const size_t smem = ((size_t)MT * AS_S + 512 * WS_S + (size_t)MT * HS_S) * 2
                      + (size_t)MT * sizeof(int);     // ~114 KB (<< 320 KB/WGP)
  dim3 grid(TOK / MT, NEXP);
  moe_ffn<<<grid, 256, smem, stream>>>(x, W1, b1, W2, b2, counts, lists, out);
}
